// PlasticityModelMoE_33672543600764
// MI455X (gfx1250) — matrix operand order repaired and checked by reading, not compiled, not run on hardware
//
#include <hip/hip_runtime.h>
#include <hip/hip_bf16.h>
#include <math.h>

// Problem constants (from reference setup_inputs)
constexpr int BB = 1024;   // batch
constexpr int DD = 1024;   // input dim
constexpr int HH = 2048;   // hidden
constexpr int EE = 8;      // experts
constexpr int MM = 2048;   // memory slots

constexpr int TILE_M = 128;
constexpr int TILE_N = 128;
constexpr int TILE_K = 32;

// LDS layouts (dword units).
// A tiles: bf16 [128][32], K-pairs packed in dwords; 16 data dwords + 4 pad per
// row -> 80B row stride (16B multiple => aligned b128 reads, no bank conflicts).
constexpr int A_ROW_DW = 20;
// B tiles: fragment-ordered; slot per (col-tile, lane) of 8 data dwords + 4 pad
// -> 48B slot stride (16B multiple).
constexpr int B_SLOT_DW = 12;

typedef __attribute__((ext_vector_type(16))) __bf16 v16bf;
typedef __attribute__((ext_vector_type(2)))  __bf16 v2bf;
typedef __attribute__((ext_vector_type(8)))  float  v8f;
typedef __attribute__((ext_vector_type(4)))  float  v4f;
typedef __attribute__((ext_vector_type(2)))  unsigned int v2u;
typedef __attribute__((ext_vector_type(4)))  unsigned int v4u;

#define ZV8 {0.f,0.f,0.f,0.f,0.f,0.f,0.f,0.f}

// Split two f32 into packed bf16 hi dword + packed bf16 residual dword.
__device__ __forceinline__ void split2(float f0, float f1,
                                       unsigned int& hi, unsigned int& lo)
{
  __bf16 h0 = (__bf16)f0, h1 = (__bf16)f1;           // v_cvt_pk_bf16_f32
  float r0 = f0 - (float)h0, r1 = f1 - (float)h1;    // v_fma_mix_f32_bf16
  v2bf hv; hv[0] = h0; hv[1] = h1;
  v2bf lv; lv[0] = (__bf16)r0; lv[1] = (__bf16)r1;
  hi = __builtin_bit_cast(unsigned int, hv);
  lo = __builtin_bit_cast(unsigned int, lv);
}

// ---------------------------------------------------------------------------
// Software-pipelined staging: global -> registers, then convert -> LDS.
// ---------------------------------------------------------------------------
struct StageRegs { v4f a[4]; v4f b[4]; };

__device__ __forceinline__ void stage_load(StageRegs& R,
    const float* __restrict__ gA, int lda, int rowBase,
    const float* __restrict__ gB, int ldb, int colBase,
    int kBase, int tid)
{
#pragma unroll
  for (int p = 0; p < 4; ++p) {                 // A: 128x32 f32 = 1024 float4
    int q = tid + p * 256;
    int r = q >> 3;                             // 0..127
    int c = (q & 7) << 2;                       // 0..28
    R.a[p] = *reinterpret_cast<const v4f*>(gA + (size_t)(rowBase + r) * lda + kBase + c);
  }
  const int K0 = (tid >> 5) * 4;                // 0..28 (K rows)
  const int c  = (tid & 31) * 4;                // 0..124 (cols)
#pragma unroll
  for (int j = 0; j < 4; ++j)                   // B: rows K0..K0+3, cols c..c+3
    R.b[j] = *reinterpret_cast<const v4f*>(gB + (size_t)(kBase + K0 + j) * ldb + colBase + c);
}

__device__ __forceinline__ void stage_store(const StageRegs& R,
    unsigned int* AsHi, unsigned int* AsLo,
    unsigned int* BsHi, unsigned int* BsLo, int tid)
{
#pragma unroll
  for (int p = 0; p < 4; ++p) {                 // A tile
    int q = tid + p * 256;
    int r = q >> 3;
    int c = (q & 7) << 2;
    int dw = r * A_ROW_DW + (c >> 1);           // even -> 8B aligned
    unsigned int h0, h1, l0, l1;
    split2(R.a[p][0], R.a[p][1], h0, l0);
    split2(R.a[p][2], R.a[p][3], h1, l1);
    v2u hv = {h0, h1}, lv = {l0, l1};
    *reinterpret_cast<v2u*>(AsHi + dw) = hv;
    *reinterpret_cast<v2u*>(AsLo + dw) = lv;
  }
  const int K0  = (tid >> 5) * 4;
  const int c   = (tid & 31) * 4;
  const int h   = (K0 >> 3) & 1;                // K 8-group parity (0/1) -> lane group
  const int vi0 = ((K0 >> 4) << 2) + ((K0 & 7) >> 1); // dword index within slot (even)
#pragma unroll
  for (int j2 = 0; j2 < 4; ++j2) {              // B tile (fragment-ordered)
    int cc   = c + j2;
    int t    = cc >> 4;                         // col tile 0..7
    int lane = (cc & 15) + 16 * h;
    int slot = (t * 32 + lane) * B_SLOT_DW + vi0;
    unsigned int h0, h1, l0, l1;
    split2(R.b[0][j2], R.b[1][j2], h0, l0);     // K pair (K0, K0+1)
    split2(R.b[2][j2], R.b[3][j2], h1, l1);     // K pair (K0+2, K0+3)
    v2u hv = {h0, h1}, lv = {l0, l1};
    *reinterpret_cast<v2u*>(BsHi + slot) = hv;
    *reinterpret_cast<v2u*>(BsLo + slot) = lv;
  }
}

// ---------------------------------------------------------------------------
// One 32-deep K chunk for a wave's 32x64 sub-tile: pure ds_load_b128 fragment
// fetch + bf16x3 WMMA (hi*hi + hi*lo + lo*hi, f32 accumulate).
// A frag (16x32 bf16): lane l: row=l&15, kb=(l>>4)*8; VGPR vi holds
//   K = kb + 2vi (+8 if vi>=4) -> two contiguous 16B runs in our LDS layout.
// B frag (32x16 bf16): lane l: col=l&15, K=(l>>4)*8 + 2vi (+8 if vi>=4) -> one 32B slot.
// C frag (16x16 f32): lane l: col=l&15, row = r + 8*(l>>4).
// ---------------------------------------------------------------------------
__device__ __forceinline__ v16bf frag_from_runs(v4u q0, v4u q1)
{
  return __builtin_bit_cast(v16bf, __builtin_shufflevector(q0, q1, 0,1,2,3,4,5,6,7));
}

__device__ __forceinline__ void mma_chunk(
    const unsigned int* AsHi, const unsigned int* AsLo,
    const unsigned int* BsHi, const unsigned int* BsLo,
    int rowOff, int colTile0, int lane, v8f acc[2][4])
{
  const int mL = lane & 15;
  const int hb = lane >> 4;
  v16bf aHi[2], aLo[2];
#pragma unroll
  for (int i = 0; i < 2; ++i) {
    int base = (rowOff + i * 16 + mL) * A_ROW_DW + hb * 4;   // mult of 4 -> 16B
    aHi[i] = frag_from_runs(*reinterpret_cast<const v4u*>(AsHi + base),
                            *reinterpret_cast<const v4u*>(AsHi + base + 8));
    aLo[i] = frag_from_runs(*reinterpret_cast<const v4u*>(AsLo + base),
                            *reinterpret_cast<const v4u*>(AsLo + base + 8));
  }
#pragma unroll
  for (int j = 0; j < 4; ++j) {
    int slot = ((colTile0 + j) * 32 + lane) * B_SLOT_DW;     // mult of 4 -> 16B
    v16bf bHi = frag_from_runs(*reinterpret_cast<const v4u*>(BsHi + slot),
                               *reinterpret_cast<const v4u*>(BsHi + slot + 4));
    v16bf bLo = frag_from_runs(*reinterpret_cast<const v4u*>(BsLo + slot),
                               *reinterpret_cast<const v4u*>(BsLo + slot + 4));
#pragma unroll
    for (int i = 0; i < 2; ++i) {
      acc[i][j] = __builtin_amdgcn_wmma_f32_16x16x32_bf16(
          false, aHi[i], false, bHi, (short)0, acc[i][j], false, false);
      acc[i][j] = __builtin_amdgcn_wmma_f32_16x16x32_bf16(
          false, aHi[i], false, bLo, (short)0, acc[i][j], false, false);
      acc[i][j] = __builtin_amdgcn_wmma_f32_16x16x32_bf16(
          false, aLo[i], false, bHi, (short)0, acc[i][j], false, false);
    }
  }
}

__device__ __forceinline__ void gemm_loop(
    const float* __restrict__ gA, int lda, int rowBase,
    const float* __restrict__ gB, int ldb, int colBase,
    int K, unsigned int* AsHi, unsigned int* AsLo,
    unsigned int* BsHi, unsigned int* BsLo,
    int tid, int lane, int rowOff, int colTile0, v8f acc[2][4])
{
  StageRegs R;
  stage_load(R, gA, lda, rowBase, gB, ldb, colBase, 0, tid);
  for (int kc = 0; kc < K; kc += TILE_K) {
    __syncthreads();                       // previous chunk's LDS reads done
    stage_store(R, AsHi, AsLo, BsHi, BsLo, tid);
    __syncthreads();
    if (kc + TILE_K < K) {                 // issue next chunk's global loads now
      stage_load(R, gA, lda, rowBase, gB, ldb, colBase, kc + TILE_K, tid);
      if (kc + 2 * TILE_K < K) {           // and nudge chunk k+2 toward L2
        __builtin_prefetch(gA + (size_t)(rowBase + (tid >> 3)) * lda + kc + 2 * TILE_K, 0, 1);
        __builtin_prefetch(gB + (size_t)(kc + 2 * TILE_K + (tid >> 5)) * ldb + colBase
                               + ((tid & 31) << 2), 0, 1);
      }
    }
    mma_chunk(AsHi, AsLo, BsHi, BsLo, rowOff, colTile0, lane, acc);
  }
}

// ---------------------------------------------------------------------------
// K1: gate = softmax(x @ gate_w) over E=8.  One block per batch row.
// ---------------------------------------------------------------------------
__global__ __launch_bounds__(256)
void gate_kernel(const float* __restrict__ x, const float* __restrict__ gate_w,
                 float* __restrict__ gate)
{
  __shared__ float sm[256 * 8 + 8];
  const int b = blockIdx.x, tid = threadIdx.x;
  float acc[8] = {0.f, 0.f, 0.f, 0.f, 0.f, 0.f, 0.f, 0.f};
  for (int d = tid; d < DD; d += 256) {
    float xv = x[(size_t)b * DD + d];
    const v4f* g = reinterpret_cast<const v4f*>(gate_w + (size_t)d * EE);
    v4f g0 = g[0], g1 = g[1];
    acc[0] += xv * g0[0]; acc[1] += xv * g0[1]; acc[2] += xv * g0[2]; acc[3] += xv * g0[3];
    acc[4] += xv * g1[0]; acc[5] += xv * g1[1]; acc[6] += xv * g1[2]; acc[7] += xv * g1[3];
  }
#pragma unroll
  for (int e = 0; e < 8; ++e) sm[tid * 8 + e] = acc[e];
  __syncthreads();
  if (tid < 8) {
    float s = 0.f;
    for (int t = 0; t < 256; ++t) s += sm[t * 8 + tid];
    sm[2048 + tid] = s;
  }
  __syncthreads();
  if (tid == 0) {
    float mx = sm[2048];
    for (int e = 1; e < 8; ++e) mx = fmaxf(mx, sm[2048 + e]);
    float ex[8], s = 0.f;
    for (int e = 0; e < 8; ++e) { ex[e] = expf(sm[2048 + e] - mx); s += ex[e]; }
    float inv = 1.f / s;
    for (int e = 0; e < 8; ++e) gate[(size_t)b * EE + e] = ex[e] * inv;
  }
}

// ---------------------------------------------------------------------------
// K2: per-expert connectivity MLP -> cm = sigmoid(h1 @ w2 + b2) * mask
// ---------------------------------------------------------------------------
__global__ __launch_bounds__(256)
void conn_kernel(const float* __restrict__ neuron_avg,
                 const float* __restrict__ conn_w1, const float* __restrict__ conn_b1,
                 const float* __restrict__ conn_w2, const float* __restrict__ conn_b2,
                 const float* __restrict__ neuron_mask, float* __restrict__ cm)
{
  __shared__ float sm[256 * 32];
  __shared__ float h1[32];
  const int e = blockIdx.x, tid = threadIdx.x;
  float acc[32];
#pragma unroll
  for (int k = 0; k < 32; ++k) acc[k] = 0.f;
  for (int h = tid; h < HH; h += 256) {
    float nv = neuron_avg[(size_t)e * HH + h];
    const float* w = conn_w1 + ((size_t)e * HH + h) * 32;
#pragma unroll
    for (int k = 0; k < 32; ++k) acc[k] += nv * w[k];
  }
#pragma unroll
  for (int k = 0; k < 32; ++k) sm[tid * 32 + k] = acc[k];
  __syncthreads();
  if (tid < 32) {
    float s = conn_b1[e * 32 + tid];
    for (int t = 0; t < 256; ++t) s += sm[t * 32 + tid];
    h1[tid] = fmaxf(s, 0.f);
  }
  __syncthreads();
  for (int h = tid; h < HH; h += 256) {
    float s = conn_b2[(size_t)e * HH + h];
#pragma unroll
    for (int k = 0; k < 32; ++k) s += h1[k] * conn_w2[((size_t)e * 32 + k) * HH + h];
    float c = 1.f / (1.f + expf(-s));
    cm[(size_t)e * HH + h] = c * neuron_mask[(size_t)e * HH + h];
  }
}

// ---------------------------------------------------------------------------
// K3: fused MoE: moe_out[b,h] = sum_e gate[b,e] * relu((x@W_e + b_e) * cm[e])
// ---------------------------------------------------------------------------
__global__ __launch_bounds__(256)
void moe_expert_kernel(const float* __restrict__ x, const float* __restrict__ expert_w,
                       const float* __restrict__ expert_b, const float* __restrict__ gate,
                       const float* __restrict__ cm, float* __restrict__ moe_out)
{
  __shared__ unsigned int AsHi[TILE_M * A_ROW_DW];
  __shared__ unsigned int AsLo[TILE_M * A_ROW_DW];
  __shared__ unsigned int BsHi[8 * 32 * B_SLOT_DW];
  __shared__ unsigned int BsLo[8 * 32 * B_SLOT_DW];
  const int tid = threadIdx.x, lane = tid & 31, wave = tid >> 5;
  const int rowOff = (wave >> 1) * 32, colOff = (wave & 1) * 64;
  const int rowBase = blockIdx.y * TILE_M, colBase = blockIdx.x * TILE_N;
  const int hiHalf = (lane >> 4) * 8, nL = lane & 15;

  v8f accM[2][4];
#pragma unroll
  for (int i = 0; i < 2; ++i)
#pragma unroll
    for (int j = 0; j < 4; ++j) { v8f z = ZV8; accM[i][j] = z; }

  for (int e = 0; e < EE; ++e) {
    v8f accZ[2][4];
#pragma unroll
    for (int i = 0; i < 2; ++i)
#pragma unroll
      for (int j = 0; j < 4; ++j) { v8f z = ZV8; accZ[i][j] = z; }

    const float* W = expert_w + (size_t)e * DD * HH;
    gemm_loop(x, DD, rowBase, W, HH, colBase, DD, AsHi, AsLo, BsHi, BsLo,
              tid, lane, rowOff, colOff >> 4, accZ);

#pragma unroll
    for (int i = 0; i < 2; ++i) {
      float g8[8];
#pragma unroll
      for (int r = 0; r < 8; ++r) {
        int brow = rowBase + rowOff + i * 16 + hiHalf + r;
        g8[r] = gate[(size_t)brow * EE + e];
      }
#pragma unroll
      for (int j = 0; j < 4; ++j) {
        int hcol = colBase + colOff + j * 16 + nL;
        float cmv = cm[(size_t)e * HH + hcol];
        float bv  = expert_b[(size_t)e * HH + hcol];
#pragma unroll
        for (int r = 0; r < 8; ++r) {
          float z = accZ[i][j][r] + bv;
          float a = fmaxf(z * cmv, 0.f);
          accM[i][j][r] += g8[r] * a;
        }
      }
    }
  }
#pragma unroll
  for (int i = 0; i < 2; ++i)
#pragma unroll
    for (int j = 0; j < 4; ++j) {
      int hcol = colBase + colOff + j * 16 + nL;
#pragma unroll
      for (int r = 0; r < 8; ++r) {
        int brow = rowBase + rowOff + i * 16 + hiHalf + r;
        moe_out[(size_t)brow * HH + hcol] = accM[i][j][r];
      }
    }
}

// ---------------------------------------------------------------------------
// K4: attn logits = moe_out @ mem_read_w + mem_read_b
// ---------------------------------------------------------------------------
__global__ __launch_bounds__(256)
void attn_logits_kernel(const float* __restrict__ moe_out,
                        const float* __restrict__ mem_read_w,
                        const float* __restrict__ mem_read_b,
                        float* __restrict__ attn)
{
  __shared__ unsigned int AsHi[TILE_M * A_ROW_DW];
  __shared__ unsigned int AsLo[TILE_M * A_ROW_DW];
  __shared__ unsigned int BsHi[8 * 32 * B_SLOT_DW];
  __shared__ unsigned int BsLo[8 * 32 * B_SLOT_DW];
  const int tid = threadIdx.x, lane = tid & 31, wave = tid >> 5;
  const int rowOff = (wave >> 1) * 32, colOff = (wave & 1) * 64;
  const int rowBase = blockIdx.y * TILE_M, colBase = blockIdx.x * TILE_N;
  const int hiHalf = (lane >> 4) * 8, nL = lane & 15;

  v8f acc[2][4];
#pragma unroll
  for (int i = 0; i < 2; ++i)
#pragma unroll
    for (int j = 0; j < 4; ++j) { v8f z = ZV8; acc[i][j] = z; }

  gemm_loop(moe_out, HH, rowBase, mem_read_w, MM, colBase, HH,
            AsHi, AsLo, BsHi, BsLo, tid, lane, rowOff, colOff >> 4, acc);

#pragma unroll
  for (int i = 0; i < 2; ++i)
#pragma unroll
    for (int j = 0; j < 4; ++j) {
      int mcol = colBase + colOff + j * 16 + nL;
      float bv = mem_read_b[mcol];
#pragma unroll
      for (int r = 0; r < 8; ++r) {
        int brow = rowBase + rowOff + i * 16 + hiHalf + r;
        attn[(size_t)brow * MM + mcol] = acc[i][j][r] + bv;
      }
    }
}

// ---------------------------------------------------------------------------
// K4b: in-place row softmax over M=2048
// ---------------------------------------------------------------------------
__global__ __launch_bounds__(256)
void softmax_row_kernel(float* __restrict__ attn)
{
  __shared__ float red[256];
  __shared__ float bc;
  const int b = blockIdx.x, tid = threadIdx.x;
  float* row = attn + (size_t)b * MM;
  float mx = -3.402823466e38f;
  for (int i = tid; i < MM; i += 256) mx = fmaxf(mx, row[i]);
  red[tid] = mx; __syncthreads();
  for (int s = 128; s > 0; s >>= 1) {
    if (tid < s) red[tid] = fmaxf(red[tid], red[tid + s]);
    __syncthreads();
  }
  if (tid == 0) bc = red[0];
  __syncthreads();
  mx = bc;
  float sum = 0.f;
  for (int i = tid; i < MM; i += 256) { float ev = expf(row[i] - mx); row[i] = ev; sum += ev; }
  red[tid] = sum; __syncthreads();
  for (int s = 128; s > 0; s >>= 1) {
    if (tid < s) red[tid] += red[tid + s];
    __syncthreads();
  }
  if (tid == 0) bc = red[0];
  __syncthreads();
  float inv = 1.f / bc;
  for (int i = tid; i < MM; i += 256) row[i] *= inv;
}

// ---------------------------------------------------------------------------
// K5: read_vec = attn @ memory; out = learned_activation(moe_out + read_vec)
// ---------------------------------------------------------------------------
__device__ __forceinline__ float learned_act(float v, const float p[8])
{
  float sig  = 1.f / (1.f + expf(-v));
  float elu  = (v > 0.f) ? v : expm1f(v);
  float th   = tanhf(v);
  float relu = fmaxf(v, 0.f);
  float silu = v * sig;
  float gelu = 0.5f * v * (1.f + erff(v * 0.70710678118654752f));
  float selu = 1.0507009873554805f * ((v > 0.f) ? v : 1.6732632423543772f * expm1f(v));
  float sp   = (v > 20.f) ? v : log1pf(expf(v));
  float mish = v * tanhf(sp);
  return p[0] * sig + p[1] * elu + p[2] * th + p[3] * relu
       + p[4] * silu + p[5] * gelu + p[6] * selu + p[7] * mish;
}

__global__ __launch_bounds__(256)
void readout_kernel(const float* __restrict__ attn, const float* __restrict__ memory,
                    const float* __restrict__ moe_out, const float* __restrict__ act_w,
                    float* __restrict__ out)
{
  __shared__ unsigned int AsHi[TILE_M * A_ROW_DW];
  __shared__ unsigned int AsLo[TILE_M * A_ROW_DW];
  __shared__ unsigned int BsHi[8 * 32 * B_SLOT_DW];
  __shared__ unsigned int BsLo[8 * 32 * B_SLOT_DW];
  const int tid = threadIdx.x, lane = tid & 31, wave = tid >> 5;
  const int rowOff = (wave >> 1) * 32, colOff = (wave & 1) * 64;
  const int rowBase = blockIdx.y * TILE_M, colBase = blockIdx.x * TILE_N;
  const int hiHalf = (lane >> 4) * 8, nL = lane & 15;

  v8f acc[2][4];
#pragma unroll
  for (int i = 0; i < 2; ++i)
#pragma unroll
    for (int j = 0; j < 4; ++j) { v8f z = ZV8; acc[i][j] = z; }

  gemm_loop(attn, MM, rowBase, memory, HH, colBase, MM,
            AsHi, AsLo, BsHi, BsLo, tid, lane, rowOff, colOff >> 4, acc);

  // softmax of 9 activation-blend weights (p[8] weights the zero 'linear' branch)
  float wv[9], p[8];
#pragma unroll
  for (int i = 0; i < 9; ++i) wv[i] = act_w[i];
  float mx = wv[0];
#pragma unroll
  for (int i = 1; i < 9; ++i) mx = fmaxf(mx, wv[i]);
  float s = 0.f;
#pragma unroll
  for (int i = 0; i < 9; ++i) { wv[i] = expf(wv[i] - mx); s += wv[i]; }
  float inv = 1.f / s;
#pragma unroll
  for (int i = 0; i < 8; ++i) p[i] = wv[i] * inv;

#pragma unroll
  for (int i = 0; i < 2; ++i)
#pragma unroll
    for (int j = 0; j < 4; ++j) {
      int hcol = colBase + colOff + j * 16 + nL;
#pragma unroll
      for (int r = 0; r < 8; ++r) {
        int brow = rowBase + rowOff + i * 16 + hiHalf + r;
        size_t idx = (size_t)brow * HH + hcol;
        float v = moe_out[idx] + acc[i][j][r];
        out[idx] = learned_act(v, p);
      }
    }
}

// ---------------------------------------------------------------------------
extern "C" void kernel_launch(void* const* d_in, const int* in_sizes, int n_in,
                              void* d_out, int out_size, void* d_ws, size_t ws_size,
                              hipStream_t stream)
{
  const float* x           = (const float*)d_in[0];
  const float* gate_w      = (const float*)d_in[1];
  const float* expert_w    = (const float*)d_in[2];
  const float* expert_b    = (const float*)d_in[3];
  const float* conn_w1     = (const float*)d_in[4];
  const float* conn_b1     = (const float*)d_in[5];
  const float* conn_w2     = (const float*)d_in[6];
  const float* conn_b2     = (const float*)d_in[7];
  const float* neuron_avg  = (const float*)d_in[8];
  const float* neuron_mask = (const float*)d_in[9];
  const float* mem_read_w  = (const float*)d_in[10];
  const float* mem_read_b  = (const float*)d_in[11];
  const float* memory      = (const float*)d_in[12];
  const float* act_w       = (const float*)d_in[13];
  float* out = (float*)d_out;

  float* ws      = (float*)d_ws;
  float* gate    = ws;                              // B*E
  float* cm      = gate + (size_t)BB * EE;          // E*H
  float* moe_out = cm + (size_t)EE * HH;            // B*H
  float* attn    = moe_out + (size_t)BB * HH;       // B*M

  gate_kernel<<<BB, 256, 0, stream>>>(x, gate_w, gate);
  conn_kernel<<<EE, 256, 0, stream>>>(neuron_avg, conn_w1, conn_b1,
                                      conn_w2, conn_b2, neuron_mask, cm);
  dim3 g3(HH / TILE_N, BB / TILE_M);
  moe_expert_kernel<<<g3, 256, 0, stream>>>(x, expert_w, expert_b, gate, cm, moe_out);
  dim3 g4(MM / TILE_N, BB / TILE_M);
  attn_logits_kernel<<<g4, 256, 0, stream>>>(moe_out, mem_read_w, mem_read_b, attn);
  softmax_row_kernel<<<BB, 256, 0, stream>>>(attn);
  dim3 g5(HH / TILE_N, BB / TILE_M);
  readout_kernel<<<g5, 256, 0, stream>>>(attn, memory, moe_out, act_w, out);
}
